// s2s_bi_ncp_81346680586934
// MI455X (gfx1250) — matrix operand order repaired and checked by reading, not compiled, not run on hardware
//
#include <hip/hip_runtime.h>
#include <hip/hip_bf16.h>

typedef __attribute__((ext_vector_type(16))) _Float16 v16h;
typedef __attribute__((ext_vector_type(8)))  _Float16 v8h;
typedef __attribute__((ext_vector_type(8)))  float    v8f;

union AFrag { v16h v; v8h h[2]; };

__device__ __forceinline__ v8f wmma_f16(v16h a, v16h b, v8f c) {
    return __builtin_amdgcn_wmma_f32_16x16x32_f16(false, a, false, b, (short)0, c, false, false);
}

// ---------------------------------------------------------------------------
// Weight packer: W (row-major [rows, fan], optional mask) -> f16 fragments in
// WMMA 16x16x32 B-operand lane layout, padded/zeroed. Two column segments:
//   padded k in [0,p0)      -> original col k        (valid if k < c0)
//   padded k in [p0,p0+c1)  -> original col c0+(k-p0)
// Fragment layout: dst[((nt*KT + kt)*32 + lane)*16 + e]
//   n = nt*16 + (lane&15);  k = kt*32 + (e<8 ? (lane>>4)*8 + e : 16 + (lane>>4)*8 + e-8)
// ---------------------------------------------------------------------------
__global__ __launch_bounds__(256) void pack_w(const float* __restrict__ W,
                                              const float* __restrict__ M,
                                              _Float16* __restrict__ dst,
                                              int rows, int fan,
                                              int c0, int p0, int c1,
                                              int KT, int NT) {
    int idx = blockIdx.x * 256 + threadIdx.x;
    int total = NT * KT * 512;
    if (idx >= total) return;
    int e    = idx & 15;
    int lane = (idx >> 4) & 31;
    int rest = idx >> 9;
    int kt   = rest % KT;
    int nt   = rest / KT;
    int n = nt * 16 + (lane & 15);
    int k = kt * 32 + (e < 8 ? (lane >> 4) * 8 + e : 16 + (lane >> 4) * 8 + (e - 8));
    float v = 0.f;
    if (n < rows) {
        int col = -1;
        if (k < p0) { if (k < c0) col = k; }
        else        { int r = k - p0; if (r < c1) col = c0 + r; }
        if (col >= 0) {
            v = W[(size_t)n * fan + col];
            if (M) v *= M[(size_t)n * fan + col];
        }
    }
    dst[idx] = (_Float16)v;
}

// ---------------------------------------------------------------------------
// One CfC layer: 4 masked linears (ff1, ff2, time_a, time_b) as WMMA GEMMs on
// a [16 x Kp] LDS activation tile, combine with tanh/sigmoid, stage to hout.
// ---------------------------------------------------------------------------
__device__ void cfc_layer(const _Float16* __restrict__ wpk,
                          const float* __restrict__ b1, const float* __restrict__ b2,
                          const float* __restrict__ ba, const float* __restrict__ bb,
                          const _Float16* __restrict__ zin, int Kp,
                          _Float16* __restrict__ hout, int Np, int h_orig,
                          int wave, int nwaves, int lane) {
    const int NT = Np >> 4;
    const int KT = Kp >> 5;
    const int linstride = NT * KT * 512;
    const int col = lane & 15;
    const int hi  = lane >> 4;               // 0 or 1: K-half select
    const _Float16* zrow = zin + col * Kp + hi * 8;   // A: row = lane&15
    for (int nt = wave; nt < NT; nt += nwaves) {
        v8f acc0 = {}, acc1 = {}, acc2 = {}, acc3 = {};
        const _Float16* wb = wpk + ((size_t)(nt * KT) * 32 + lane) * 16;
        for (int kt = 0; kt < KT; ++kt) {
            AFrag a;
            a.h[0] = *(const v8h*)(zrow + kt * 32);
            a.h[1] = *(const v8h*)(zrow + kt * 32 + 16);
            const _Float16* wp = wb + (size_t)kt * 512;
            AFrag f0, f1, f2, f3;
            f0.h[0] = *(const v8h*)(wp);                   f0.h[1] = *(const v8h*)(wp + 8);
            f1.h[0] = *(const v8h*)(wp + linstride);       f1.h[1] = *(const v8h*)(wp + linstride + 8);
            f2.h[0] = *(const v8h*)(wp + 2 * linstride);   f2.h[1] = *(const v8h*)(wp + 2 * linstride + 8);
            f3.h[0] = *(const v8h*)(wp + 3 * linstride);   f3.h[1] = *(const v8h*)(wp + 3 * linstride + 8);
            acc0 = wmma_f16(a.v, f0.v, acc0);
            acc1 = wmma_f16(a.v, f1.v, acc1);
            acc2 = wmma_f16(a.v, f2.v, acc2);
            acc3 = wmma_f16(a.v, f3.v, acc3);
        }
        int n = nt * 16 + col;
        bool valid = n < h_orig;
        float c1 = valid ? b1[n] : 0.f;
        float c2 = valid ? b2[n] : 0.f;
        float ca = valid ? ba[n] : 0.f;
        float cb = valid ? bb[n] : 0.f;
        int m0 = hi * 8;
#pragma unroll
        for (int r = 0; r < 8; ++r) {
            float f1v = tanhf(acc0[r] + c1);
            float f2v = tanhf(acc1[r] + c2);
            float tt  = acc2[r] + acc3[r] + ca + cb;
            float sg  = 1.f / (1.f + __expf(-tt));
            float hv  = f1v + sg * (f2v - f1v);
            hout[(m0 + r) * Np + n] = (_Float16)hv;
        }
    }
}

// Copy staged layer output into recurrent slot (self) and next layer's input.
__device__ void scatter_h(const _Float16* __restrict__ hout, int Np,
                          _Float16* __restrict__ zself, int selfKp, int self_off,
                          _Float16* __restrict__ znext, int nextKp,
                          int tid, int nthreads) {
    int total = 16 * Np;
    for (int i = tid; i < total; i += nthreads) {
        int m = i / Np, c = i - m * Np;
        _Float16 v = hout[i];
        zself[m * selfKp + self_off + c] = v;
        if (znext) znext[m * nextKp + c] = v;
    }
}

// ---------------------------------------------------------------------------
// Encoder: blocks 0..15 = forward scan (512 steps), 16..31 = backward (1 step:
// only out_b[:, -1, :] == backward-scan step 0 is consumed downstream).
// Padded dims: L1 z=[x 64 | h1 80] Kp=160 Np=80; L2 z=[h1 80 | h2 64] Kp=160
// Np=64; L3 z=[h2 64 | h3 128] Kp=192 Np=128.
// ---------------------------------------------------------------------------
struct EncArgs {
    const float* x;
    const _Float16* wpk_f;
    const _Float16* wpk_b;
    const float* bias_f[12];   // [layer*4 + lin]
    const float* bias_b[12];
    _Float16* enc_out;         // [256][512] f16 (fwd cols 0..256, bwd 256..512)
};

__global__ __launch_bounds__(256) void enc_kernel(EncArgs a) {
    __shared__ __align__(16) _Float16 z1[16 * 160];
    __shared__ __align__(16) _Float16 z2[16 * 160];
    __shared__ __align__(16) _Float16 z3[16 * 192];
    __shared__ __align__(16) _Float16 hout[16 * 192];
    const int tid = threadIdx.x, lane = tid & 31, wave = tid >> 5;
    const int dir = blockIdx.x >> 4;
    const int b0  = (blockIdx.x & 15) * 16;

    for (int i = tid; i < 16 * 160; i += 256) { z1[i] = (_Float16)0.f; z2[i] = (_Float16)0.f; }
    for (int i = tid; i < 16 * 192; i += 256) { z3[i] = (_Float16)0.f; }
    __syncthreads();

    const _Float16* wpk = dir ? a.wpk_b : a.wpk_f;
    const float* bp[12];
#pragma unroll
    for (int i = 0; i < 12; ++i) bp[i] = dir ? a.bias_b[i] : a.bias_f[i];

    const int steps = dir ? 1 : 512;
    for (int s = 0; s < steps; ++s) {
        const int t = dir ? 511 : s;
        for (int i = tid; i < 16 * 64; i += 256) {
            int m = i >> 6, c = i & 63;
            z1[m * 160 + c] = (_Float16)a.x[((size_t)(b0 + m) * 512 + t) * 64 + c];
        }
        __syncthreads();
        cfc_layer(wpk,          bp[0], bp[1], bp[2],  bp[3],  z1, 160, hout,  80,  77, wave, 8, lane);
        __syncthreads();
        scatter_h(hout, 80, z1, 160, 64, z2, 160, tid, 256);
        __syncthreads();
        cfc_layer(wpk + 51200,  bp[4], bp[5], bp[6],  bp[7],  z2, 160, hout,  64,  51, wave, 8, lane);
        __syncthreads();
        scatter_h(hout, 64, z2, 160, 80, z3, 192, tid, 256);
        __syncthreads();
        cfc_layer(wpk + 92160,  bp[8], bp[9], bp[10], bp[11], z3, 192, hout, 128, 128, wave, 8, lane);
        __syncthreads();
        scatter_h(hout, 128, z3, 192, 64, (_Float16*)nullptr, 0, tid, 256);
        __syncthreads();
    }
    // Final concat(h1[:77], h2[:51], h3[:128]) -> enc_out
    for (int i = tid; i < 16 * 256; i += 256) {
        int m = i >> 8, c = i & 255;
        _Float16 v;
        if (c < 77)       v = z1[m * 160 + 64 + c];
        else if (c < 128) v = z2[m * 160 + 80 + (c - 77)];
        else              v = z3[m * 192 + 64 + (c - 128)];
        a.enc_out[(size_t)(b0 + m) * 512 + dir * 256 + c] = v;
    }
}

// hidden[B,256] = enc_out[B,512] @ Wfc^T + bfc   (16 blocks x 16 waves)
__global__ __launch_bounds__(512) void fc_kernel(const _Float16* __restrict__ enc_out,
                                                 const _Float16* __restrict__ fc_pk,
                                                 const float* __restrict__ bfc,
                                                 _Float16* __restrict__ hidden) {
    const int tid = threadIdx.x, lane = tid & 31, wave = tid >> 5;
    const int b0 = blockIdx.x * 16;
    v8f acc = {};
    const _Float16* arow = enc_out + (size_t)(b0 + (lane & 15)) * 512 + (lane >> 4) * 8;
    for (int kt = 0; kt < 16; ++kt) {
        AFrag af, bf;
        af.h[0] = *(const v8h*)(arow + kt * 32);
        af.h[1] = *(const v8h*)(arow + kt * 32 + 16);
        const _Float16* wp = fc_pk + ((size_t)(wave * 16 + kt) * 32 + lane) * 16;
        bf.h[0] = *(const v8h*)wp; bf.h[1] = *(const v8h*)(wp + 8);
        acc = wmma_f16(af.v, bf.v, acc);
    }
    const int n = wave * 16 + (lane & 15);
    const float bb = bfc[n];
    const int m0 = (lane >> 4) * 8;
#pragma unroll
    for (int r = 0; r < 8; ++r)
        hidden[(size_t)(b0 + m0 + r) * 256 + n] = (_Float16)(acc[r] + bb);
}

// ---------------------------------------------------------------------------
// Decoder: 512 autoregressive steps. Padded dims: L1 z=[y 16 | h1 160] Kp=192
// Np=160; L2 z=[h1 160 | h2 112] Kp=288 Np=112; L3 z=[h2 112 | h3 16] Kp=128
// Np=16. Prediction via WMMA with dec_fc packed to N=16 (col 0 valid).
// ---------------------------------------------------------------------------
struct DecArgs {
    const _Float16* wpk;
    const _Float16* dfc_pk;
    const _Float16* hidden;    // [256][256] f16
    const float* bias[12];
    const float* bd;
    float* out;                // [256][512]
};

__global__ __launch_bounds__(256) void dec_kernel(DecArgs a) {
    __shared__ __align__(16) _Float16 z1[16 * 192];
    __shared__ __align__(16) _Float16 z2[16 * 288];
    __shared__ __align__(16) _Float16 z3[16 * 128];
    __shared__ __align__(16) _Float16 hout[16 * 160];
    __shared__ __align__(16) _Float16 cbuf[16 * 256];
    const int tid = threadIdx.x, lane = tid & 31, wave = tid >> 5;
    const int b0 = blockIdx.x * 16;

    // init recurrent slots from hidden = split([153,102,1])
    for (int i = tid; i < 16 * 192; i += 256) {
        int m = i / 192, c = i - m * 192;
        _Float16 v = (_Float16)0.f;
        if (c >= 16 && c < 16 + 153) v = a.hidden[(size_t)(b0 + m) * 256 + (c - 16)];
        z1[i] = v;
    }
    for (int i = tid; i < 16 * 288; i += 256) {
        int m = i / 288, c = i - m * 288;
        _Float16 v = (_Float16)0.f;
        if (c >= 160 && c < 160 + 102) v = a.hidden[(size_t)(b0 + m) * 256 + 153 + (c - 160)];
        z2[i] = v;
    }
    for (int i = tid; i < 16 * 128; i += 256) {
        int m = i / 128, c = i - m * 128;
        _Float16 v = (_Float16)0.f;
        if (c == 112) v = a.hidden[(size_t)(b0 + m) * 256 + 255];
        z3[i] = v;
    }
    __syncthreads();

    for (int s = 0; s < 512; ++s) {
        cfc_layer(a.wpk,          a.bias[0], a.bias[1], a.bias[2],  a.bias[3],  z1, 192, hout, 160, 153, wave, 8, lane);
        __syncthreads();
        scatter_h(hout, 160, z1, 192, 16, z2, 288, tid, 256);
        __syncthreads();
        cfc_layer(a.wpk + 122880, a.bias[4], a.bias[5], a.bias[6],  a.bias[7],  z2, 288, hout, 112, 102, wave, 8, lane);
        __syncthreads();
        scatter_h(hout, 112, z2, 288, 160, z3, 128, tid, 256);
        __syncthreads();
        cfc_layer(a.wpk + 251904, a.bias[8], a.bias[9], a.bias[10], a.bias[11], z3, 128, hout, 16, 1, wave, 8, lane);
        __syncthreads();
        scatter_h(hout, 16, z3, 128, 112, (_Float16*)nullptr, 0, tid, 256);
        __syncthreads();
        // cbuf = concat(h1[:153], h2[:102], h3[:1])
        for (int i = tid; i < 16 * 256; i += 256) {
            int m = i >> 8, c = i & 255;
            _Float16 v;
            if (c < 153)      v = z2[m * 288 + c];
            else if (c < 255) v = z3[m * 128 + (c - 153)];
            else              v = z3[m * 128 + 112];
            cbuf[i] = v;
        }
        __syncthreads();
        if (wave == 0) {
            v8f acc = {};
            const _Float16* arow = cbuf + (lane & 15) * 256 + (lane >> 4) * 8;
            for (int kt = 0; kt < 8; ++kt) {
                AFrag af, bf;
                af.h[0] = *(const v8h*)(arow + kt * 32);
                af.h[1] = *(const v8h*)(arow + kt * 32 + 16);
                const _Float16* wp = a.dfc_pk + ((size_t)kt * 32 + lane) * 16;
                bf.h[0] = *(const v8h*)wp; bf.h[1] = *(const v8h*)(wp + 8);
                acc = wmma_f16(af.v, bf.v, acc);
            }
            if ((lane & 15) == 0) {
                const float bd = a.bd[0];
                const int m0 = (lane >> 4) * 8;
#pragma unroll
                for (int r = 0; r < 8; ++r) {
                    float p = acc[r] + bd;
                    a.out[(size_t)(b0 + m0 + r) * 512 + s] = p;   // preds feed d_out
                    z1[(m0 + r) * 192 + 0] = (_Float16)p;          // y feedback
                }
            }
        }
        __syncthreads();
    }
}

// ---------------------------------------------------------------------------
// Host side
// ---------------------------------------------------------------------------
extern "C" void kernel_launch(void* const* d_in, const int* in_sizes, int n_in,
                              void* d_out, int out_size, void* d_ws, size_t ws_size,
                              hipStream_t stream) {
    (void)in_sizes; (void)n_in; (void)out_size; (void)ws_size;
    auto F = [&](int i) { return (const float*)d_in[i]; };
    // leaf order: x | enc_f (3 layers x (W1,b1,W2,b2,Wa,ba,Wb,bb)) | enc_b |
    //             enc_fc (W,b) | dec | dec_fc (W,b) | masks f/b/dec
    const int X = 0, ENCF = 1, ENCB = 25, FCW = 49, FCB = 50, DEC = 51,
              DFCW = 75, DFCB = 76, MF = 77, MB = 80, MD = 83;

    _Float16* ws      = (_Float16*)d_ws;
    _Float16* wpk_f   = ws;
    _Float16* wpk_b   = ws + 190464;
    _Float16* wpk_d   = ws + 380928;
    _Float16* fc_pk   = ws + 641024;
    _Float16* dfc_pk  = ws + 772096;
    _Float16* enc_out = ws + 776192;
    _Float16* hidden  = ws + 907264;

    struct LP { int rows, fan, c0, p0, KT, NT; };
    const LP encL[3] = {{77, 141, 64, 64, 5, 5}, {51, 128, 77, 80, 5, 4}, {128, 179, 51, 64, 6, 8}};
    const LP decL[3] = {{153, 154, 1, 16, 6, 10}, {102, 255, 153, 160, 9, 7}, {1, 103, 102, 112, 4, 1}};

    auto pack = [&](const float* W, const float* M, _Float16* dst, const LP& p) {
        int total = p.NT * p.KT * 512;
        pack_w<<<(total + 255) / 256, 256, 0, stream>>>(W, M, dst, p.rows, p.fan,
                                                        p.c0, p.p0, p.rows, p.KT, p.NT);
    };
    // masked CfC weights (enc_f, enc_b, dec)
    {
        size_t off = 0;
        for (int l = 0; l < 3; ++l) {
            size_t lin = (size_t)encL[l].NT * encL[l].KT * 512;
            for (int j = 0; j < 4; ++j) {
                pack(F(ENCF + l * 8 + j * 2), F(MF + l), wpk_f + off + j * lin, encL[l]);
                pack(F(ENCB + l * 8 + j * 2), F(MB + l), wpk_b + off + j * lin, encL[l]);
            }
            off += 4 * lin;
        }
    }
    {
        size_t off = 0;
        for (int l = 0; l < 3; ++l) {
            size_t lin = (size_t)decL[l].NT * decL[l].KT * 512;
            for (int j = 0; j < 4; ++j)
                pack(F(DEC + l * 8 + j * 2), F(MD + l), wpk_d + off + j * lin, decL[l]);
            off += 4 * lin;
        }
    }
    // dense fc layers (no mask, no recurrent segment)
    pack_w<<<(16 * 16 * 512 + 255) / 256, 256, 0, stream>>>(F(FCW), nullptr, fc_pk,
                                                            256, 512, 512, 512, 0, 16, 16);
    pack_w<<<(8 * 512 + 255) / 256, 256, 0, stream>>>(F(DFCW), nullptr, dfc_pk,
                                                      1, 256, 256, 256, 0, 8, 1);

    // encoder (fwd 512 steps + bwd 1 step)
    EncArgs ea;
    ea.x = F(X);
    ea.wpk_f = wpk_f; ea.wpk_b = wpk_b;
    for (int l = 0; l < 3; ++l)
        for (int j = 0; j < 4; ++j) {
            ea.bias_f[l * 4 + j] = F(ENCF + l * 8 + j * 2 + 1);
            ea.bias_b[l * 4 + j] = F(ENCB + l * 8 + j * 2 + 1);
        }
    ea.enc_out = enc_out;
    enc_kernel<<<32, 256, 0, stream>>>(ea);

    fc_kernel<<<16, 512, 0, stream>>>(enc_out, fc_pk, F(FCB), hidden);

    DecArgs da;
    da.wpk = wpk_d; da.dfc_pk = dfc_pk; da.hidden = hidden;
    for (int l = 0; l < 3; ++l)
        for (int j = 0; j < 4; ++j)
            da.bias[l * 4 + j] = F(DEC + l * 8 + j * 2 + 1);
    da.bd = F(DFCB);
    da.out = (float*)d_out;
    dec_kernel<<<16, 256, 0, stream>>>(da);
}
